// POSDecoder_51230369906919
// MI455X (gfx1250) — matrix operand order repaired and checked by reading, not compiled, not run on hardware
//
#include <hip/hip_runtime.h>
#include <hip/hip_bf16.h>
#include <math.h>

// ---------------------------------------------------------------------------
// GRU POS decoder on gfx1250 (MI455X), bf16 WMMA + async LDS copies.
//   B=256 T=512 H=1024 E=128 L=17
// ---------------------------------------------------------------------------

typedef __bf16 bf16_t;
typedef __attribute__((ext_vector_type(16))) __bf16 v16bf;
typedef __attribute__((ext_vector_type(8)))  float  v8f;

#define B_    256
#define T_    512
#define H_    1024
#define E_    128
#define L_    17
#define HE_   1152      // H + E
#define G3_   3072      // 3*H
#define START_ 17

// ---- gfx1250 async global->LDS copy (ASYNCcnt path), with sync fallback ----
#if defined(__gfx1250__) && \
    __has_builtin(__builtin_amdgcn_global_load_async_to_lds_b128) && \
    __has_builtin(__builtin_amdgcn_s_wait_asynccnt)
#define HAVE_ASYNC 1
typedef int v4i_ __attribute__((vector_size(16)));
typedef __attribute__((address_space(1))) v4i_ v4i_as1;   // global
typedef __attribute__((address_space(3))) v4i_ v4i_as3;   // LDS
#define ASYNC_CP16(gsrc, ldst)                                              \
    __builtin_amdgcn_global_load_async_to_lds_b128(                         \
        (v4i_as1*)(gsrc), (v4i_as3*)(ldst), 0, 0)
#define WAIT_ASYNC() __builtin_amdgcn_s_wait_asynccnt(0)
#else
#define HAVE_ASYNC 0
#define ASYNC_CP16(gsrc, ldst) (*(uint4*)(ldst) = *(const uint4*)(gsrc))
#define WAIT_ASYNC() ((void)0)
#endif

union Frag16 {               // one WMMA bf16 operand per lane: 16 bf16 = 32B
    v16bf  v;
    uint4  q[2];
    __bf16 e[16];
};

__device__ __forceinline__ float sigmoid_f(float x){
    return 1.0f / (1.0f + __expf(-x));
}

__device__ __forceinline__ v8f wmma_bf16(const Frag16& a, const Frag16& b, v8f c){
    return __builtin_amdgcn_wmma_f32_16x16x32_bf16(
        /*neg_a=*/false, a.v, /*neg_b=*/false, b.v,
        /*c_mod=*/(short)0, c, /*reuse_a=*/false, /*reuse_b=*/false);
}

// ---------------------------------------------------------------------------
// fp32 -> bf16 weight conversion
// ---------------------------------------------------------------------------
__global__ void k_f32_to_bf16(const float* __restrict__ src,
                              bf16_t* __restrict__ dst, int n){
    int i = blockIdx.x * blockDim.x + threadIdx.x;
    if (i < n) dst[i] = (bf16_t)src[i];
}

// W_out [17,1024] -> padded bf16 [32,1024] (rows 17..31 zero)
__global__ void k_pad_wout(const float* __restrict__ W, bf16_t* __restrict__ dst){
    int i = blockIdx.x * blockDim.x + threadIdx.x;
    if (i >= 32 * H_) return;
    int n = i / H_;
    dst[i] = (n < L_) ? (bf16_t)W[n * H_ + (i % H_)] : (bf16_t)0.0f;
}

// ---------------------------------------------------------------------------
// Build x = [seg , emb(prev_label)] in bf16, row-major [B*T, 1152]
// ---------------------------------------------------------------------------
__global__ void k_build_x(const float* __restrict__ seg,
                          const int*   __restrict__ labels,
                          const float* __restrict__ emb,
                          bf16_t* __restrict__ x){
    long i = (long)blockIdx.x * blockDim.x + threadIdx.x;
    long n = (long)B_ * T_ * HE_;
    if (i >= n) return;
    int  j  = (int)(i % HE_);
    long bt = i / HE_;
    int  t  = (int)(bt % T_);
    int  b  = (int)(bt / T_);
    float v;
    if (j < H_) {
        v = seg[bt * H_ + j];
    } else {
        int prev = (t == 0) ? START_ : labels[b * T_ + t - 1];
        v = emb[prev * E_ + (j - H_)];
    }
    x[i] = (bf16_t)v;
}

// ---------------------------------------------------------------------------
// x_proj = x @ W_ih^T + b_ih    [131072 x 1152] * [1152 x 3072] -> f32
// Block: 256 thr (8 waves), tile 128(M) x 64(N); wave = 32x32 (2x2 WMMA).
// A double-buffered through LDS via async copies; B straight from L2-resident
// bf16 W_ih.
// ---------------------------------------------------------------------------
__global__ __launch_bounds__(256)
void k_xproj(const bf16_t* __restrict__ x, const bf16_t* __restrict__ Wih,
             const float* __restrict__ bih, float* __restrict__ xp){
    __shared__ alignas(16) bf16_t As[2][128 * 32];    // 2 x 8 KB ping-pong
    const int m0  = blockIdx.x * 128;
    const int n0  = blockIdx.y * 64;
    const int tid = threadIdx.x;
    const int w = tid >> 5, lane = tid & 31;
    const int wm = w >> 1, wn = w & 1;                // wave's 32x32 sub-tile
    const int lrow = lane & 15;
    const int lk   = lane >> 4;                       // lane-half (K split)

    // per-thread staging coords: 2 x 16B chunks of the 128x32 tile
    const int c0 = tid * 2;
    const int row0 = c0 >> 2,       off0 = (c0 & 3) * 8;
    const int row1 = (c0 + 1) >> 2, off1 = ((c0 + 1) & 3) * 8;
    const bf16_t* g0 = x + (long)(m0 + row0) * HE_ + off0;
    const bf16_t* g1 = x + (long)(m0 + row1) * HE_ + off1;

    // prologue: stage K-slice 0 into buffer 0
    ASYNC_CP16(g0, &As[0][row0 * 32 + off0]);
    ASYNC_CP16(g1, &As[0][row1 * 32 + off1]);

    v8f acc[2][2] = {};

    for (int kk = 0; kk < HE_; kk += 32){
        const int buf = (kk >> 5) & 1;
        WAIT_ASYNC();                 // my copies into As[buf] landed
        __syncthreads();              // everyone's copies landed / reads done
        if (kk + 32 < HE_){           // prefetch next slice into other buffer
            ASYNC_CP16(g0 + kk + 32, &As[buf ^ 1][row0 * 32 + off0]);
            ASYNC_CP16(g1 + kk + 32, &As[buf ^ 1][row1 * 32 + off1]);
        }

        Frag16 a[2], bfr[2];
        #pragma unroll
        for (int mf = 0; mf < 2; ++mf){               // A: lane=row, K-chunks
            const int r  = wm * 32 + mf * 16 + lrow;
            const int kb = lk * 8;
            a[mf].q[0] = *(const uint4*)(&As[buf][r * 32 + kb]);
            a[mf].q[1] = *(const uint4*)(&As[buf][r * 32 + kb + 16]);
        }
        #pragma unroll
        for (int nf = 0; nf < 2; ++nf){               // B: lane=col, K-chunks
            const int nn = n0 + wn * 32 + nf * 16 + lrow;
            const bf16_t* bp = Wih + (long)nn * HE_ + kk + lk * 8;
            bfr[nf].q[0] = *(const uint4*)(bp);
            bfr[nf].q[1] = *(const uint4*)(bp + 16);
        }
        #pragma unroll
        for (int mf = 0; mf < 2; ++mf)
            #pragma unroll
            for (int nf = 0; nf < 2; ++nf)
                acc[mf][nf] = wmma_bf16(a[mf], bfr[nf], acc[mf][nf]);
    }

    // epilogue: C layout = VGPR v -> M = v + 8*(lane>=16), N = lane&15
    #pragma unroll
    for (int mf = 0; mf < 2; ++mf)
        #pragma unroll
        for (int nf = 0; nf < 2; ++nf){
            int col  = n0 + wn * 32 + nf * 16 + lrow;
            float bias = bih[col];
            #pragma unroll
            for (int v = 0; v < 8; ++v){
                int row = m0 + wm * 32 + mf * 16 + v + lk * 8;
                xp[(long)row * G3_ + col] = acc[mf][nf][v] + bias;
            }
        }
}

// ---------------------------------------------------------------------------
// One GRU time step: gh = h_{t-1} @ W_hh^T (3 gates via WMMA), gate math,
// h_t -> hs (f32, for the z-blend) and hb (bf16, next step's A operand).
// Block tile: 64(batch) x 64(hidden), 8 waves, wave = 16x32 x 3 gates.
// A (= h_{t-1} bf16 shadow) double-buffered via async LDS copies.
// ---------------------------------------------------------------------------
__global__ __launch_bounds__(256)
void k_gru_step(const bf16_t* __restrict__ Whh, const float* __restrict__ xp,
                const float* __restrict__ bhh, float* __restrict__ hs,
                bf16_t* __restrict__ hb, int t){
    __shared__ alignas(16) bf16_t As[2][64 * 32];     // 2 x 4 KB ping-pong
    const int m0  = blockIdx.x * 64;                  // batch offset
    const int n0  = blockIdx.y * 64;                  // hidden offset
    const int tid = threadIdx.x;
    const int w = tid >> 5, lane = tid & 31;
    const int wm = w >> 1;                            // 0..3 -> 16-row group
    const int wn = w & 1;                             // 0..1 -> 32-col group
    const int lrow = lane & 15;
    const int lk   = lane >> 4;

    v8f accR[2] = {}, accZ[2] = {}, accN[2] = {};

    if (t > 0){                                       // t==0: h=0 => gh=0
        // per-thread staging coords: one 16B chunk of the 64x32 tile
        const int row = tid >> 2, off = (tid & 3) * 8;
        const bf16_t* gsrc = hb + (long)((m0 + row) * T_ + t - 1) * H_ + off;

        ASYNC_CP16(gsrc, &As[0][row * 32 + off]);

        for (int kk = 0; kk < H_; kk += 32){
            const int buf = (kk >> 5) & 1;
            WAIT_ASYNC();
            __syncthreads();
            if (kk + 32 < H_)
                ASYNC_CP16(gsrc + kk + 32, &As[buf ^ 1][row * 32 + off]);

            Frag16 a;
            {
                int r  = wm * 16 + lrow;
                int kb = lk * 8;
                a.q[0] = *(const uint4*)(&As[buf][r * 32 + kb]);
                a.q[1] = *(const uint4*)(&As[buf][r * 32 + kb + 16]);
            }
            #pragma unroll
            for (int nf = 0; nf < 2; ++nf){
                int j = n0 + wn * 32 + nf * 16 + lrow;
                #pragma unroll
                for (int g = 0; g < 3; ++g){          // r, z, n gate rows of W_hh
                    const bf16_t* bp = Whh + (long)(g * H_ + j) * H_ + kk + lk * 8;
                    Frag16 bfr;
                    bfr.q[0] = *(const uint4*)(bp);
                    bfr.q[1] = *(const uint4*)(bp + 16);
                    if      (g == 0) accR[nf] = wmma_bf16(a, bfr, accR[nf]);
                    else if (g == 1) accZ[nf] = wmma_bf16(a, bfr, accZ[nf]);
                    else             accN[nf] = wmma_bf16(a, bfr, accN[nf]);
                }
            }
        }
    }

    // gate math + state update (fp32)
    #pragma unroll
    for (int nf = 0; nf < 2; ++nf){
        int j = n0 + wn * 32 + nf * 16 + lrow;
        float bR = bhh[j], bZ = bhh[H_ + j], bN = bhh[2 * H_ + j];
        #pragma unroll
        for (int v = 0; v < 8; ++v){
            int  b  = m0 + wm * 16 + v + lk * 8;
            long bt = (long)b * T_ + t;
            float xr = xp[bt * G3_ + j];
            float xz = xp[bt * G3_ + H_ + j];
            float xn = xp[bt * G3_ + 2 * H_ + j];
            float r  = sigmoid_f(xr + accR[nf][v] + bR);
            float z  = sigmoid_f(xz + accZ[nf][v] + bZ);
            float nn = tanhf(xn + r * (accN[nf][v] + bN));
            float hp = (t == 0) ? 0.0f : hs[(bt - 1) * H_ + j];
            float h  = (1.0f - z) * nn + z * hp;
            hs[bt * H_ + j] = h;
            hb[bt * H_ + j] = (bf16_t)h;              // bf16 shadow for t+1
        }
    }
}

// ---------------------------------------------------------------------------
// logits = hb @ W_out^T + b_out  (W_out padded to 32 rows for WMMA N-dim)
// Each wave owns a 16-row tile, computes 16x32, stores cols < 17.
// ---------------------------------------------------------------------------
__global__ __launch_bounds__(256)
void k_outproj(const bf16_t* __restrict__ hb, const bf16_t* __restrict__ WoutP,
               const float* __restrict__ bout, float* __restrict__ out){
    const int tid = threadIdx.x;
    const int w = tid >> 5, lane = tid & 31;
    const int lrow = lane & 15, lk = lane >> 4;
    const long m0 = (long)blockIdx.x * 128 + w * 16;

    v8f acc[2] = {};
    for (int kk = 0; kk < H_; kk += 32){
        Frag16 a;
        const bf16_t* ap = hb + (m0 + lrow) * H_ + kk + lk * 8;
        a.q[0] = *(const uint4*)(ap);
        a.q[1] = *(const uint4*)(ap + 16);
        #pragma unroll
        for (int nf = 0; nf < 2; ++nf){
            const bf16_t* bp = WoutP + (long)(nf * 16 + lrow) * H_ + kk + lk * 8;
            Frag16 bfr;
            bfr.q[0] = *(const uint4*)(bp);
            bfr.q[1] = *(const uint4*)(bp + 16);
            acc[nf] = wmma_bf16(a, bfr, acc[nf]);
        }
    }
    #pragma unroll
    for (int nf = 0; nf < 2; ++nf){
        int n = nf * 16 + lrow;
        if (n < L_){
            float bias = bout[n];
            #pragma unroll
            for (int v = 0; v < 8; ++v){
                long row = m0 + v + lk * 8;
                out[row * L_ + n] = acc[nf][v] + bias;
            }
        }
    }
}

// ---------------------------------------------------------------------------
extern "C" void kernel_launch(void* const* d_in, const int* in_sizes, int n_in,
                              void* d_out, int out_size, void* d_ws, size_t ws_size,
                              hipStream_t stream){
    const float* seg    = (const float*)d_in[0];
    const int*   labels = (const int*)  d_in[1];
    const float* emb    = (const float*)d_in[2];
    const float* Wih    = (const float*)d_in[3];
    const float* Whh    = (const float*)d_in[4];
    const float* bih    = (const float*)d_in[5];
    const float* bhh    = (const float*)d_in[6];
    const float* Wout   = (const float*)d_in[7];
    const float* bout   = (const float*)d_in[8];
    float* out = (float*)d_out;

    char* ws = (char*)d_ws;
    bf16_t* Wih_b = (bf16_t*)ws;  ws += (size_t)G3_ * HE_ * 2;
    bf16_t* Whh_b = (bf16_t*)ws;  ws += (size_t)G3_ * H_  * 2;
    bf16_t* WoutP = (bf16_t*)ws;  ws += (size_t)32  * H_  * 2;
    bf16_t* x     = (bf16_t*)ws;  ws += (size_t)B_ * T_ * HE_ * 2;
    float*  xp    = (float*) ws;  ws += (size_t)B_ * T_ * G3_ * 4;
    float*  hs    = (float*) ws;  ws += (size_t)B_ * T_ * H_  * 4;
    bf16_t* hb    = (bf16_t*)ws;  ws += (size_t)B_ * T_ * H_  * 2;

    {
        int n = G3_ * HE_;
        k_f32_to_bf16<<<(n + 255) / 256, 256, 0, stream>>>(Wih, Wih_b, n);
    }
    {
        int n = G3_ * H_;
        k_f32_to_bf16<<<(n + 255) / 256, 256, 0, stream>>>(Whh, Whh_b, n);
    }
    k_pad_wout<<<(32 * H_ + 255) / 256, 256, 0, stream>>>(Wout, WoutP);
    {
        long n = (long)B_ * T_ * HE_;
        k_build_x<<<(unsigned)((n + 255) / 256), 256, 0, stream>>>(seg, labels, emb, x);
    }
    dim3 g1((B_ * T_) / 128, G3_ / 64);
    k_xproj<<<g1, 256, 0, stream>>>(x, Wih_b, bih, xp);

    dim3 g2(B_ / 64, H_ / 64);
    for (int t = 0; t < T_; ++t)
        k_gru_step<<<g2, 256, 0, stream>>>(Whh_b, xp, bhh, hs, hb, t);

    k_outproj<<<(B_ * T_) / 128, 256, 0, stream>>>(hb, WoutP, bout, out);
}
